// STDPConv2d_78632261255788
// MI455X (gfx1250) — matrix operand order repaired and checked by reading, not compiled, not run on hardware
//
#include <hip/hip_runtime.h>

typedef __attribute__((ext_vector_type(16))) _Float16 v16h;
typedef __attribute__((ext_vector_type(8)))  _Float16 v8h;
typedef __attribute__((ext_vector_type(8)))  float    v8f;

#define N_BATCH   32
#define C_IN      64
#define H_IN      64
#define W_IN      64
#define C_OUT     256
#define H_OUT     60
#define W_OUT     60
#define K_GEMM    (C_IN * 25)          // 1600
#define K_STEPS   (K_GEMM / 32)        // 50
#define N_WINNERS 5
#define INHIB_R   2
#define POT_PER   (C_OUT * H_OUT * W_OUT)   // 921600
#define OUT_TOTAL (N_BATCH * POT_PER)       // 29491200

#define LDS_BUF_BYTES 8192             // A tile (4KB) + B tile (4KB)

// ---------------------------------------------------------------------------
// f32 -> f16 pack
// ---------------------------------------------------------------------------
__global__ void cvt_f32_to_f16(const float* __restrict__ src,
                               _Float16* __restrict__ dst, int n) {
  int i = blockIdx.x * blockDim.x + threadIdx.x;
  if (i < n) dst[i] = (_Float16)src[i];
}

// ---------------------------------------------------------------------------
// Implicit-GEMM conv via v_wmma_f32_16x16x32_f16, double-buffered LDS.
// Grid: (mblk=4, h=60, nb=32), block = 128 threads = 4 waves.
// Workgroup computes a 64(M=cout) x 64(N=w, 60 valid) block of pot for one
// (batch, output-row). Wave wv owns M strip [wv*16, wv*16+16) x all 64 N
// (4 accumulators, A fragment reused across 4 WMMAs per K-step).
// A (weights) tile is DMA'd with GLOBAL_LOAD_ASYNC_TO_LDS_B128 (ASYNCcnt);
// the NEXT tile's DMA is issued before computing the CURRENT tile, so the
// async engine overlaps WMMA execution (ping-pong buffers).
// ---------------------------------------------------------------------------
__global__ __launch_bounds__(128)
void conv_wmma(const _Float16* __restrict__ W16,
               const _Float16* __restrict__ X16,
               float* __restrict__ pot) {
  extern __shared__ char smem[];       // 2 x (Alds 4KB | Blds 4KB)

  const int mblk = blockIdx.x;           // 0..3
  const int h    = blockIdx.y;           // 0..59
  const int nb   = blockIdx.z;           // 0..31
  const int tid  = threadIdx.x;
  const int wv   = tid >> 5;             // wave 0..3 -> M sub-strip
  const int lane = tid & 31;
  const int m0   = mblk * 64;

  // fragment-assembly coordinates (documented CDNA5 16-bit layouts)
  const int lm    = lane & 15;           // A: M mod 16 / B: N mod 16
  const int lhalf = lane >> 4;           // half-wave selector
  const int akb   = lhalf * 8;           // A K-base (halves)
  const int bkb   = lhalf * 8;           // B K-base (halves)
  const int arow  = wv * 16 + lm;        // this lane's A row in the 64-row tile

  // B-staging coordinates
  const int kr = tid & 31;               // k row 0..31
  const int q  = tid >> 5;               // n quarter 0..3

  // stage tile `ks` into buffer `buf`
  auto stage = [&](int ks, int buf) {
    const int k0 = ks * 32;
    // async A: 64x32 halves = 256 x 16B chunks, 2 per thread
    {
      const int c0 = tid;
      const int c1 = tid + 128;
      const _Float16* g0 =
          W16 + (size_t)(m0 + (c0 >> 2)) * K_GEMM + k0 + (c0 & 3) * 8;
      const _Float16* g1 =
          W16 + (size_t)(m0 + (c1 >> 2)) * K_GEMM + k0 + (c1 & 3) * 8;
      unsigned l0 = (unsigned)(buf * LDS_BUF_BYTES + c0 * 16);
      unsigned l1 = (unsigned)(buf * LDS_BUF_BYTES + c1 * 16);
      asm volatile(
          "global_load_async_to_lds_b128 %0, %1, off\n\t"
          "global_load_async_to_lds_b128 %2, %3, off"
          :: "v"(l0), "v"(g0), "v"(l1), "v"(g1)
          : "memory");
    }
    // scalar im2col B: 32(k) x 64(n) halves, [n][k] layout
    {
      _Float16* Bl = (_Float16*)(smem + buf * LDS_BUF_BYTES + 4096);
      const int k   = k0 + kr;
      const int cin = k / 25;
      const int r   = k % 25;
      const int kh  = r / 5;
      const int kw  = r % 5;
      const _Float16* xrow =
          X16 + ((size_t)(nb * C_IN + cin) * H_IN + (h + kh)) * W_IN + kw;
      __builtin_prefetch(xrow + W_IN, 0, 3);
#pragma unroll
      for (int j = 0; j < 16; ++j) {
        const int n = q * 16 + j;
        Bl[n * 32 + kr] = (n < W_OUT) ? xrow[n] : (_Float16)0.0f;
      }
    }
  };

  v8f acc[4] = {{}, {}, {}, {}};

  stage(0, 0);  // prologue

  for (int ks = 0; ks < K_STEPS; ++ks) {
    const int buf = ks & 1;

    // tile `ks` DMA complete + all waves' B stores visible
    asm volatile("s_wait_asynccnt 0" ::: "memory");
    __syncthreads();

    // issue tile ks+1 into the other buffer; DMA overlaps compute below
    if (ks + 1 < K_STEPS) stage(ks + 1, buf ^ 1);

    const _Float16* Alds = (const _Float16*)(smem + buf * LDS_BUF_BYTES);
    const _Float16* Blds =
        (const _Float16*)(smem + buf * LDS_BUF_BYTES + 4096);

    // A fragment (2 x ds_load_b128), reused for 4 WMMAs
    v8h alo = *reinterpret_cast<const v8h*>(&Alds[arow * 32 + akb]);
    v8h ahi = *reinterpret_cast<const v8h*>(&Alds[arow * 32 + akb + 16]);
    v16h a = __builtin_shufflevector(alo, ahi, 0, 1, 2, 3, 4, 5, 6, 7,
                                     8, 9, 10, 11, 12, 13, 14, 15);

#pragma unroll
    for (int nt = 0; nt < 4; ++nt) {
      const int nl = nt * 16 + lm;
      v8h blo = *reinterpret_cast<const v8h*>(&Blds[nl * 32 + bkb]);
      v8h bhi = *reinterpret_cast<const v8h*>(&Blds[nl * 32 + bkb + 16]);
      v16h b = __builtin_shufflevector(blo, bhi, 0, 1, 2, 3, 4, 5, 6, 7,
                                       8, 9, 10, 11, 12, 13, 14, 15);
      acc[nt] = __builtin_amdgcn_wmma_f32_16x16x32_f16(
          false, a, false, b, (short)0, acc[nt], false, false);
    }
    __syncthreads();  // protect buffer `buf` until tile ks+2 staging
  }

  // ---- write D: lane holds M = wv*16 + v + 8*lhalf, N = nt*16 + lm ----
#pragma unroll
  for (int nt = 0; nt < 4; ++nt) {
    const int w = nt * 16 + lm;
    if (w < W_OUT) {
#pragma unroll
      for (int v = 0; v < 8; ++v) {
        const int m = m0 + wv * 16 + v + 8 * lhalf;
        pot[(((size_t)nb * C_OUT + m) * H_OUT + h) * W_OUT + w] = acc[nt][v];
      }
    }
  }
}

// ---------------------------------------------------------------------------
// Iterative 5-winner argmax with spatial inhibition. One workgroup per batch
// sample; pot lives in d_out (118 MB -> L2-resident on MI455X's 192 MB L2).
// ---------------------------------------------------------------------------
__global__ __launch_bounds__(1024)
void kwta(float* __restrict__ pot_all, int* __restrict__ winners) {
  const int nb = blockIdx.x;
  float* pot = pot_all + (size_t)nb * POT_PER;

  __shared__ float sval[1024];
  __shared__ int   sidx[1024];
  __shared__ int   swin;

  const int tid = threadIdx.x;
  const float NEG_INF = -__builtin_inff();

  for (int it = 0; it < N_WINNERS; ++it) {
    float best = NEG_INF;
    int bidx = 0x7fffffff;
    for (int i = tid; i < POT_PER; i += 1024) {
      float v = pot[i];
      if (v > best || (v == best && i < bidx)) { best = v; bidx = i; }
    }
    sval[tid] = best;
    sidx[tid] = bidx;
    __syncthreads();

    for (int s = 512; s > 0; s >>= 1) {
      if (tid < s) {
        float ov = sval[tid + s];
        int   oi = sidx[tid + s];
        if (ov > sval[tid] || (ov == sval[tid] && oi < sidx[tid])) {
          sval[tid] = ov;
          sidx[tid] = oi;
        }
      }
      __syncthreads();
    }
    if (tid == 0) {
      const int wloc = (sval[0] > 0.0f) ? sidx[0] : -1;
      swin = wloc;
      winners[nb * N_WINNERS + it] = wloc;
    }
    __syncthreads();

    const int widx = swin;
    if (widx >= 0) {
      const int hw = widx % (H_OUT * W_OUT);
      const int wh = hw / W_OUT;
      const int ww = hw % W_OUT;
      const int h0 = (wh - INHIB_R > 0) ? wh - INHIB_R : 0;
      const int h1 = (wh + INHIB_R < H_OUT - 1) ? wh + INHIB_R : H_OUT - 1;
      const int w0 = (ww - INHIB_R > 0) ? ww - INHIB_R : 0;
      const int w1 = (ww + INHIB_R < W_OUT - 1) ? ww + INHIB_R : W_OUT - 1;
      const int nh = h1 - h0 + 1, nw = w1 - w0 + 1;
      const int total = C_OUT * nh * nw;
      for (int t = tid; t < total; t += 1024) {
        const int c  = t / (nh * nw);
        const int rr = t % (nh * nw);
        const int hh = h0 + rr / nw;
        const int cw = w0 + rr % nw;
        pot[(c * H_OUT + hh) * W_OUT + cw] = NEG_INF;
      }
    }
    __threadfence();
    __syncthreads();
  }
}

// ---------------------------------------------------------------------------
// finalize: binary spike map
// ---------------------------------------------------------------------------
__global__ void zero_out(float* __restrict__ out, int n) {
  int i = blockIdx.x * blockDim.x + threadIdx.x;
  if (i < n) out[i] = 0.0f;
}

__global__ void set_winners(const int* __restrict__ winners,
                            float* __restrict__ out) {
  int i = blockIdx.x * blockDim.x + threadIdx.x;
  if (i < N_BATCH * N_WINNERS) {
    const int widx = winners[i];
    if (widx >= 0)
      out[(size_t)(i / N_WINNERS) * POT_PER + widx] = 1.0f;
  }
}

// ---------------------------------------------------------------------------
extern "C" void kernel_launch(void* const* d_in, const int* in_sizes, int n_in,
                              void* d_out, int out_size, void* d_ws,
                              size_t ws_size, hipStream_t stream) {
  const float* x      = (const float*)d_in[0];   // (32,64,64,64)
  const float* weight = (const float*)d_in[1];   // (256,64,5,5)
  float* out = (float*)d_out;                    // (32,256,60,60)

  // workspace layout: [winners 640B | pad to 1KB | W16 f16 | X16 f16]
  char* ws = (char*)d_ws;
  int* winners  = (int*)ws;
  _Float16* W16 = (_Float16*)(ws + 1024);                 // 16B-aligned
  const int    w_elems = C_OUT * K_GEMM;                  // 409600
  _Float16* X16 = (_Float16*)(ws + 1024 + (size_t)w_elems * sizeof(_Float16));
  const size_t x_elems = (size_t)N_BATCH * C_IN * H_IN * W_IN;  // 8388608

  cvt_f32_to_f16<<<(w_elems + 255) / 256, 256, 0, stream>>>(
      weight, W16, w_elems);
  cvt_f32_to_f16<<<(int)((x_elems + 255) / 256), 256, 0, stream>>>(
      x, X16, (int)x_elems);

  conv_wmma<<<dim3(4, 60, 32), 128, 2 * LDS_BUF_BYTES, stream>>>(
      W16, X16, out);

  kwta<<<N_BATCH, 1024, 0, stream>>>(out, winners);

  zero_out<<<(OUT_TOTAL + 255) / 256, 256, 0, stream>>>(out, OUT_TOTAL);
  set_winners<<<1, 256, 0, stream>>>(winners, out);
}
